// AttentionTSSA_65738769433253
// MI455X (gfx1250) — matrix operand order repaired and checked by reading, not compiled, not run on hardware
//
#include <hip/hip_runtime.h>
#include <hip/hip_bf16.h>

typedef __attribute__((ext_vector_type(16))) _Float16 v16h;
typedef __attribute__((ext_vector_type(8)))  float    v8f;
typedef __attribute__((ext_vector_type(2)))  _Float16 h2;

constexpr int B_ = 64, C_ = 256, T_ = 64, V_ = 25, H_ = 8;
constexpr int N_ = T_ * V_;      // 1600
constexpr int MT = C_ / 16;      // 16 output-row tiles
constexpr int NT = N_ / 16;      // 100 output-col tiles
constexpr int NB = 4;            // N-tiles per wave (register blocking)
constexpr int NTB = NT / NB;     // 25 blocked column groups

// ---------------- f32 -> f16 weight convert ----------------
__global__ __launch_bounds__(256) void cvt_f16(const float* __restrict__ s,
                                               _Float16* __restrict__ d, int n) {
    int i = blockIdx.x * 256 + threadIdx.x;
    if (i < n) d[i] = (_Float16)s[i];
}

// ---------------- pack x [B][C][N] f32 -> xh [B][N][C] f16 ----------------
__global__ __launch_bounds__(256) void pack_xT(const float* __restrict__ x,
                                               _Float16* __restrict__ xh) {
    size_t idx = (size_t)blockIdx.x * 256 + threadIdx.x;   // (b, c, n), n fastest
    int n = (int)(idx % N_);
    size_t t2 = idx / N_;
    int c = (int)(t2 % C_);
    int b = (int)(t2 / C_);
    xh[((size_t)b * N_ + n) * C_ + c] = (_Float16)x[idx];
}

// ---------------- WMMA GEMM with 4-way N register blocking -------------------
// Wh:  [C_][C_] f16 row-major (O x K)
// Xt:  [B_][N_][C_] f16 (per position n, a K-vector of channels)
// Each wave: one 16-row M tile x four 16-col N tiles; A fragment reused 4x.
// All fragment loads for a K-step are issued before the 4 WMMAs so the
// scheduler can use staggered partial s_wait_loadcnt instead of full drains.
// OUTMODE 0: out[b][n][o] f32  (feeds the scan kernel, coalesced on channels)
// OUTMODE 1: out[b][o][n] f32 + bias[o]  (final output layout [B,C,T,V])
template <int OUTMODE>
__global__ __launch_bounds__(256) void wmma_gemm(const _Float16* __restrict__ Wh,
                                                 const _Float16* __restrict__ Xt,
                                                 const float* __restrict__ bias,
                                                 float* __restrict__ out) {
    int wave = (blockIdx.x * 256 + (int)threadIdx.x) >> 5;
    int lane = threadIdx.x & 31;
    int half = lane >> 4;     // K-half selector for A/B frags, M-half for D
    int lm   = lane & 15;     // M row (A) / N col (B, D)

    int ntb = wave % NTB;
    int mt  = (wave / NTB) % MT;
    int b   = wave / (NTB * MT);
    int n0  = ntb * (16 * NB);

    const _Float16* arow = Wh + (size_t)(mt * 16 + lm) * C_;
    const _Float16* brow = Xt + ((size_t)b * N_ + n0 + lm) * C_;

    v8f acc[NB];
#pragma unroll
    for (int j = 0; j < NB; ++j) acc[j] = (v8f){};

#pragma unroll 2
    for (int k0 = 0; k0 < C_; k0 += 32) {
        // 16-bit A 16x32 layout: K = 16*(v>>2) + 8*half + 2*(v&3) (+slot)
        v16h a;
#pragma unroll
        for (int v = 0; v < 8; ++v) {
            int ka = k0 + ((v & 4) << 2) + half * 8 + ((v & 3) << 1);
            h2 av = *(const h2*)(arow + ka);
            a[2 * v]     = av.x;
            a[2 * v + 1] = av.y;
        }
        // 16-bit B 32x16 layout: K = 16*(v>>2) + 8*half + 2*(v&3) (+slot), N = lm
        v16h bm[NB];
#pragma unroll
        for (int j = 0; j < NB; ++j) {
#pragma unroll
            for (int v = 0; v < 8; ++v) {
                int kb = k0 + ((v & 4) << 2) + half * 8 + ((v & 3) << 1);
                h2 bv = *(const h2*)(brow + (size_t)j * 16 * C_ + kb);
                bm[j][2 * v]     = bv.x;
                bm[j][2 * v + 1] = bv.y;
            }
        }
#pragma unroll
        for (int j = 0; j < NB; ++j) {
            acc[j] = __builtin_amdgcn_wmma_f32_16x16x32_f16(false, a, false, bm[j],
                                                            (short)0, acc[j],
                                                            false, false);
        }
    }

    // D 16x16 f32: VGPR r -> M = r + 8*half, N = lm
#pragma unroll
    for (int j = 0; j < NB; ++j) {
#pragma unroll
        for (int r = 0; r < 8; ++r) {
            int o = mt * 16 + r + 8 * half;
            int n = n0 + j * 16 + lm;
            if (OUTMODE == 0) {
                out[((size_t)b * N_ + n) * C_ + o] = acc[j][r];
            } else {
                out[((size_t)b * C_ + o) * N_ + n] = acc[j][r] + bias[o];
            }
        }
    }
}

// ---------------- sequential-scan middle stage -------------------------------
// One block per batch b. Wave h owns head h, lane owns d (d==warpSize==32).
// w: [B][N][C] f32 ; writes yh: [B][N][C] f16.
__global__ __launch_bounds__(256) void tssa_middle(const float* __restrict__ w,
                                                   const float* __restrict__ temp,
                                                   const float* __restrict__ dbias,
                                                   _Float16* __restrict__ yh) {
    __shared__ float pi[H_ * N_];   // 8*1600*4 = 51200 B
    int b   = blockIdx.x;
    int tid = threadIdx.x;
    int h   = tid >> 5;
    int d   = tid & 31;

    const float* wb = w + (size_t)b * N_ * C_;
    float db = dbias[h];
    float tp = temp[h];

    // pass 1: denom cumsum + per-(h,n) logits
    float cum = 0.f;
    for (int n = 0; n < N_; ++n) {
        if (n + 8 < N_) __builtin_prefetch(wb + (size_t)(n + 8) * C_ + tid, 0, 3);
        float wv  = wb[(size_t)n * C_ + tid];
        float wsq = wv * wv;
        cum += wsq;
        float wn = wsq / fmaxf(cum, 1e-12f) + db;   // +db per d element (broadcast)
#pragma unroll
        for (int off = 16; off >= 1; off >>= 1)
            wn += __shfl_xor(wn, off, 32);
        if (d == 0) pi[h * N_ + n] = wn * tp;
    }
    __syncthreads();

    // pass 2: softmax over heads at each n (in-place in LDS)
    for (int n = tid; n < N_; n += 256) {
        float t[H_];
        float m = -1e30f;
#pragma unroll
        for (int hh = 0; hh < H_; ++hh) { t[hh] = pi[hh * N_ + n]; m = fmaxf(m, t[hh]); }
        float s = 0.f;
#pragma unroll
        for (int hh = 0; hh < H_; ++hh) { t[hh] = __expf(t[hh] - m); s += t[hh]; }
        float inv = 1.f / s;
#pragma unroll
        for (int hh = 0; hh < H_; ++hh) pi[hh * N_ + n] = t[hh] * inv;
    }
    __syncthreads();

    // pass 3: weighted cumsum ratio -> y (f16 for GEMM2)
    float cn = 0.f, cd = 0.f;
    _Float16* yb = yh + (size_t)b * N_ * C_;
    for (int n = 0; n < N_; ++n) {
        if (n + 8 < N_) __builtin_prefetch(wb + (size_t)(n + 8) * C_ + tid, 0, 3);
        float wv  = wb[(size_t)n * C_ + tid];
        float wsq = wv * wv;
        float P   = pi[h * N_ + n];
        cn += wsq * P;
        cd += P;
        float dots = cn / (cd + 1e-8f);
        float attn = 1.f / (1.f + dots);
        yb[(size_t)n * C_ + tid] = (_Float16)(-(wv * P) * attn);
    }
}

extern "C" void kernel_launch(void* const* d_in, const int* in_sizes, int n_in,
                              void* d_out, int out_size, void* d_ws, size_t ws_size,
                              hipStream_t stream) {
    const float* x     = (const float*)d_in[0];
    const float* Wqkv  = (const float*)d_in[1];
    const float* temp  = (const float*)d_in[2];
    const float* dbias = (const float*)d_in[3];
    const float* Wproj = (const float*)d_in[4];
    const float* bproj = (const float*)d_in[5];
    float* out = (float*)d_out;

    // workspace carve-up (~158 MB): xh reused as yh after GEMM1 consumes it
    char* ws = (char*)d_ws;
    _Float16* xh = (_Float16*)ws;                       // [B][N][C] f16, 52.4 MB
    ws += (size_t)B_ * N_ * C_ * sizeof(_Float16);
    float* wbuf = (float*)ws;                           // [B][N][C] f32, 104.9 MB
    ws += (size_t)B_ * N_ * C_ * sizeof(float);
    _Float16* Wqkvh = (_Float16*)ws;                    // 128 KB
    ws += (size_t)C_ * C_ * sizeof(_Float16);
    _Float16* Wprojh = (_Float16*)ws;                   // 128 KB
    _Float16* yh = xh;                                  // reuse

    cvt_f16<<<dim3((C_ * C_ + 255) / 256), dim3(256), 0, stream>>>(Wqkv, Wqkvh, C_ * C_);
    cvt_f16<<<dim3((C_ * C_ + 255) / 256), dim3(256), 0, stream>>>(Wproj, Wprojh, C_ * C_);

    size_t tot = (size_t)B_ * C_ * N_;                  // 26,214,400 (multiple of 256)
    pack_xT<<<dim3((unsigned)(tot / 256)), dim3(256), 0, stream>>>(x, xh);

    int gemm_blocks = (B_ * MT * NTB) / 8;              // 3200 blocks x 8 waves
    wmma_gemm<0><<<dim3(gemm_blocks), dim3(256), 0, stream>>>(Wqkvh, xh, bproj, wbuf);

    tssa_middle<<<dim3(B_), dim3(256), 0, stream>>>(wbuf, temp, dbias, yh);

    wmma_gemm<1><<<dim3(gemm_blocks), dim3(256), 0, stream>>>(Wprojh, yh, bproj, out);
}
